// surrogate_70669391889119
// MI455X (gfx1250) — matrix operand order repaired and checked by reading, not compiled, not run on hardware
//
#include <hip/hip_runtime.h>
#include <hip/hip_bf16.h>
#include <math.h>
#include <stdint.h>

typedef __attribute__((ext_vector_type(16))) _Float16 v16h;
typedef __attribute__((ext_vector_type(8)))  _Float16 v8h;
typedef __attribute__((ext_vector_type(8)))  float    v8f;

union V16 { v16h v; v8h h[2]; };

// ----------------------------------------------------------------------------
// CDNA5 async global->LDS copy (ASYNCcnt-tracked) + waits   [ISA 10 / 15.18.3]
// lds_off is the wave-relative LDS byte address (generic LDS aperture keeps it
// in addr[31:0], so truncating a generic __shared__ pointer is exact).
// ----------------------------------------------------------------------------
__device__ __forceinline__ void async_b128(unsigned lds_off, const _Float16* g) {
    asm volatile("global_load_async_to_lds_b128 %0, %1, off"
                 :: "v"(lds_off), "v"(g)
                 : "memory");
}
__device__ __forceinline__ void wait_async_le3() {
    asm volatile("s_wait_asynccnt 0x3" ::: "memory");
}
__device__ __forceinline__ void wait_async_0() {
    asm volatile("s_wait_asynccnt 0x0" ::: "memory");
}

// ----------------------------------------------------------------------------
// scalar prep: reg = exp(reg_param)+1e-8, t = exp(t_param)+1e-8
// ----------------------------------------------------------------------------
__global__ void k_scalars(const float* __restrict__ regp,
                          const float* __restrict__ tp,
                          float* __restrict__ sc) {
    if (threadIdx.x == 0) {
        sc[0] = expf(regp[0]) + 1e-8f;  // reg
        sc[1] = expf(tp[0])   + 1e-8f;  // t
    }
}

// ----------------------------------------------------------------------------
// elementwise f32 -> f16
// ----------------------------------------------------------------------------
__global__ void k_f32_to_f16(const float* __restrict__ src,
                             _Float16* __restrict__ dst, size_t n) {
    size_t i = (size_t)blockIdx.x * blockDim.x + threadIdx.x;
    if (i < n) dst[i] = (_Float16)src[i];
}

// ----------------------------------------------------------------------------
// embedding gather: tok[Nrow][L] -> dst[Nrow][L*D] (f16), rows of we[V][D]
// ----------------------------------------------------------------------------
__global__ void k_gather(const int* __restrict__ tok, const float* __restrict__ we,
                         _Float16* __restrict__ dst, int Nrow, int L, int D) {
    size_t i = (size_t)blockIdx.x * blockDim.x + threadIdx.x;
    size_t total = (size_t)Nrow * L * D;
    if (i >= total) return;
    int d = (int)(i % D);
    size_t nl = i / D;
    int l = (int)(nl % L);
    int n = (int)(nl / L);
    int t = tok[(size_t)n * L + l];
    dst[i] = (_Float16)we[(size_t)t * D + d];
}

// ----------------------------------------------------------------------------
// WMMA NT-GEMM:  C[M,N] = act(A[M,K] * B[N,K]^T + bias[N])
// block = 256 threads = 8 wave32s (4 M-waves x 2 N-waves); block tile 64Mx128N.
// K-step 32; A(64x32) and B(128x32) tiles double-buffered in LDS, filled with
// global_load_async_to_lds_b128 (3 async ops / wave / stage), so the next
// stage's DMA overlaps this stage's 4 v_wmma_f32_16x16x32_f16 per wave.
// M%64==0, N%128==0, K%32==0 (true for all GEMMs here).
// ----------------------------------------------------------------------------
template <bool RELU>
__global__ __launch_bounds__(256) void k_gemm_nt(
    const _Float16* __restrict__ A, const _Float16* __restrict__ B,
    const float* __restrict__ bias,
    _Float16* __restrict__ Ch, float* __restrict__ Cf,
    int M, int N, int K)
{
    (void)M;
    constexpr int ROWB   = 80;               // 64B of K-data + 16B bank pad
    constexpr int ABYTES = 64 * ROWB;        // 5120
    constexpr int BBYTES = 128 * ROWB;       // 10240
    constexpr int BUFB   = ABYTES + BBYTES;  // 15360
    __shared__ __align__(16) char smem[2 * BUFB];

    const int tid  = threadIdx.x;
    const int lane = tid & 31;
    const int wid  = tid >> 5;
    const int wm   = wid & 3;
    const int wn   = wid >> 2;
    const int r    = lane & 15;
    const bool hi  = lane >= 16;

    // ---- async fill addressing: thread -> (row = tid/4, 16B chunk = tid%4) ----
    const int frow = tid >> 2;               // 0..63
    const int fchk = (tid & 3) * 16;         // byte offset in row
    const _Float16* ga  = A + (size_t)(blockIdx.x * 64  + frow) * K + (tid & 3) * 8;
    const _Float16* gb0 = B + (size_t)(blockIdx.y * 128 + frow) * K + (tid & 3) * 8;
    const _Float16* gb1 = gb0 + (size_t)64 * K;
    const unsigned lbase = (unsigned)(uintptr_t)&smem[0];
    const unsigned la  = lbase + frow * ROWB + fchk;
    const unsigned lb0 = lbase + ABYTES + frow * ROWB + fchk;
    const unsigned lb1 = lb0 + 64 * ROWB;

    // ---- WMMA fragment offsets in LDS                         [ISA 7.12.2] ----
    // A 16x32: lane holds row (wm*16+r); chunks at K{0..7,16..23} (lo lanes)
    // or K{8..15,24..31} (hi lanes) -> bytes (hi?16:0) and +32.
    const int aoff  = (wm * 16 + r) * ROWB + (hi ? 16 : 0);
    // B 32x16: lane holds W-row (wn*64 + t*16 + r), K{0..7,16..23}(lo)/K{8..15,24..31}(hi)
    // -> bytes (hi?16:0) and +32.
    const int boff0 = ABYTES + (wn * 64 + r) * ROWB + (hi ? 16 : 0);

    auto issue = [&](int kk, int buf) {
        const unsigned o = (unsigned)(buf * BUFB);
        async_b128(la  + o, ga  + kk);
        async_b128(lb0 + o, gb0 + kk);
        async_b128(lb1 + o, gb1 + kk);
    };

    v8f acc[4] = {};
    issue(0, 0);
    int buf = 0;
    for (int k0 = 0; k0 < K; k0 += 32) {
        if (k0 + 32 < K) { issue(k0 + 32, buf ^ 1); wait_async_le3(); }
        else             { wait_async_0(); }
        __syncthreads();                     // stage data visible to all waves

        const char* sb = &smem[buf * BUFB];
        V16 a;
        a.h[0] = *(const v8h*)(sb + aoff);
        a.h[1] = *(const v8h*)(sb + aoff + 32);
#pragma unroll
        for (int t = 0; t < 4; ++t) {
            const char* bp = sb + boff0 + t * 16 * ROWB;
            V16 b;
            b.h[0] = *(const v8h*)(bp);
            b.h[1] = *(const v8h*)(bp + 32);
            acc[t] = __builtin_amdgcn_wmma_f32_16x16x32_f16(
                false, a.v, false, b.v, (short)0, acc[t], false, false);
        }
        __syncthreads();   // all reads of this buffer done before next overwrite
        buf ^= 1;
    }

    // C layout: VGPR v -> row bm + v + (hi?8:0); col = bn + t*16 + r
    const int bm = blockIdx.x * 64 + wm * 16;
    const int bn = blockIdx.y * 128 + wn * 64;
#pragma unroll
    for (int t = 0; t < 4; ++t) {
        int col = bn + t * 16 + r;
        float bs = bias ? bias[col] : 0.f;
#pragma unroll
        for (int v = 0; v < 8; ++v) {
            int row = bm + v + (hi ? 8 : 0);
            float val = acc[t][v] + bs;
            if (RELU) val = fmaxf(val, 0.f);
            size_t o = (size_t)row * N + col;
            if (Ch) Ch[o] = (_Float16)val;
            if (Cf) Cf[o] = val;
        }
    }
}

// ----------------------------------------------------------------------------
// per-column std (ddof=1) + 0.1  -> xmul[C]
// ----------------------------------------------------------------------------
__global__ void k_colstd(const float* __restrict__ X, float* __restrict__ xmul,
                         int N, int C) {
    int c = blockIdx.x * blockDim.x + threadIdx.x;
    if (c >= C) return;
    float s = 0.f, s2 = 0.f;
    for (int n = 0; n < N; ++n) {
        float v = X[(size_t)n * C + c];
        s += v; s2 += v * v;
    }
    float var = (s2 - s * s / (float)N) / (float)(N - 1);
    xmul[c] = sqrtf(fmaxf(var, 0.f)) + 0.1f;
}

// X[n][c] /= xmul[c] (in place, workspace buffer)
__global__ void k_scale_cols(float* __restrict__ X, const float* __restrict__ xmul,
                             int N, int C) {
    size_t i = (size_t)blockIdx.x * blockDim.x + threadIdx.x;
    if (i >= (size_t)N * C) return;
    X[i] /= xmul[i % C];
}

// ----------------------------------------------------------------------------
// Y stats: ymul (std ddof=1 + 0.1), b = mean(Y/ymul), yc2[j] = sum_n Yc^2
// ----------------------------------------------------------------------------
__global__ void k_ystats(const float* __restrict__ Y, float* __restrict__ ymul,
                         float* __restrict__ bvec, float* __restrict__ yc2,
                         int N, int NY) {
    __shared__ float ss[16][17], ss2[16][17];
    int j  = threadIdx.x % 16;
    int rr = threadIdx.x / 16;
    float s = 0.f, s2 = 0.f;
    for (int n = rr; n < N; n += 16) {
        float v = Y[(size_t)n * NY + j];
        s += v; s2 += v * v;
    }
    ss[rr][j] = s; ss2[rr][j] = s2;
    __syncthreads();
    if (threadIdx.x < 16) {
        int jj = threadIdx.x;
        float S = 0.f, S2 = 0.f;
        for (int r = 0; r < 16; ++r) { S += ss[r][jj]; S2 += ss2[r][jj]; }
        float var = (S2 - S * S / (float)N) / (float)(N - 1);
        float ym  = sqrtf(fmaxf(var, 0.f)) + 0.1f;
        float b   = S / ((float)N * ym);
        ymul[jj] = ym;
        bvec[jj] = b;
        yc2[jj]  = S2 / (ym * ym) - (float)N * b * b;   // sum (Y/ym - b)^2
    }
}

// ----------------------------------------------------------------------------
// amat[f][j] = sum_n Xn[n][f] * (Y[n][j]/ymul[j] - b[j])    (Xn normalized)
// ----------------------------------------------------------------------------
__global__ void k_amat(const float* __restrict__ Xn, const float* __restrict__ Y,
                       const float* __restrict__ ymul, const float* __restrict__ bvec,
                       float* __restrict__ amat, int N, int C, int NY) {
    size_t id = (size_t)blockIdx.x * blockDim.x + threadIdx.x;
    if (id >= (size_t)C * NY) return;
    int j = (int)(id % NY);
    int f = (int)(id / NY);
    float ym = ymul[j], b = bvec[j];
    float acc = 0.f;
    for (int n = 0; n < N; ++n)
        acc += Xn[(size_t)n * C + f] * (Y[(size_t)n * NY + j] / ym - b);
    amat[id] = acc;
}

// ----------------------------------------------------------------------------
// per-group covariance: covi[g] = t * Xg^T Xg + reg*I   (Xg = Xn[:, g*64:+64])
// ----------------------------------------------------------------------------
__global__ __launch_bounds__(256) void k_cov(const float* __restrict__ Xn,
                                             const float* __restrict__ sc,
                                             float* __restrict__ covi,
                                             int N, int C) {
    int g = blockIdx.x;
    __shared__ float tile[64][65];
    float reg = sc[0], tpar = sc[1];
    int i  = threadIdx.x >> 2;             // 0..63
    int j0 = (threadIdx.x & 3) * 16;       // 0,16,32,48
    float acc[16];
#pragma unroll
    for (int q = 0; q < 16; ++q) acc[q] = 0.f;

    for (int nb = 0; nb < N; nb += 64) {
        __syncthreads();
        for (int e = threadIdx.x; e < 64 * 64; e += 256) {
            int rr = e >> 6, cc = e & 63;
            tile[rr][cc] = Xn[(size_t)(nb + rr) * C + g * 64 + cc];
        }
        __syncthreads();
        for (int rr = 0; rr < 64; ++rr) {
            float xi = tile[rr][i];
#pragma unroll
            for (int q = 0; q < 16; ++q) acc[q] += xi * tile[rr][j0 + q];
        }
    }
#pragma unroll
    for (int q = 0; q < 16; ++q) {
        int j = j0 + q;
        covi[((size_t)g * 64 + i) * 64 + j] = tpar * acc[q] + (i == j ? reg : 0.f);
    }
}

// ----------------------------------------------------------------------------
// fp64 Gauss-Jordan inversion of covi[g] (64x64, SPD) + slogdet via pivots.
// ----------------------------------------------------------------------------
__global__ __launch_bounds__(64) void k_inv(const float* __restrict__ covi,
                                            float* __restrict__ cov,
                                            float* __restrict__ Sg) {
    int g = blockIdx.x;
    int t = threadIdx.x;
    __shared__ double Aug[64][128];
    for (int j = 0; j < 64; ++j) {
        Aug[t][j]      = (double)covi[((size_t)g * 64 + t) * 64 + j];
        Aug[t][64 + j] = (t == j) ? 1.0 : 0.0;
    }
    double logdet = 0.0;
    for (int k = 0; k < 64; ++k) {
        __syncthreads();
        double piv = Aug[k][k];
        if (t == 0) logdet += log(fabs(piv));
        if (t == k) {
            double ip = 1.0 / piv;
            for (int j = 0; j < 128; ++j) Aug[k][j] *= ip;
        }
        __syncthreads();
        if (t != k) {
            double f = Aug[t][k];
            for (int j = 0; j < 128; ++j) Aug[t][j] -= f * Aug[k][j];
        }
    }
    __syncthreads();
    for (int j = 0; j < 64; ++j)
        cov[((size_t)g * 64 + t) * 64 + j] = (float)Aug[t][64 + j];
    if (t == 0) Sg[g] = (float)logdet;
}

// ----------------------------------------------------------------------------
// w0[g] = t * cov[g] @ a[g]  (64x16);  s2[g][j] = sum_i w0 * (cov @ w0)
// ----------------------------------------------------------------------------
__global__ __launch_bounds__(256) void k_w0(const float* __restrict__ cov,
                                            const float* __restrict__ amat,
                                            const float* __restrict__ sc,
                                            float* __restrict__ w0,
                                            float* __restrict__ s2) {
    int g = blockIdx.x;
    __shared__ float cs[64][65];
    __shared__ float as[64][16];
    __shared__ float wsld[64][16];
    __shared__ float s2s[16];
    float tpar = sc[1];
    for (int e = threadIdx.x; e < 64 * 64; e += 256)
        cs[e >> 6][e & 63] = cov[(size_t)g * 4096 + e];
    for (int e = threadIdx.x; e < 64 * 16; e += 256)
        as[e >> 4][e & 15] = amat[(size_t)g * 1024 + e];
    if (threadIdx.x < 16) s2s[threadIdx.x] = 0.f;
    __syncthreads();

    float myw[4];
#pragma unroll
    for (int q = 0; q < 4; ++q) {
        int e = threadIdx.x + 256 * q;
        int i = e >> 4, j = e & 15;
        float acc = 0.f;
        for (int k = 0; k < 64; ++k) acc += cs[i][k] * as[k][j];
        myw[q] = tpar * acc;
        wsld[i][j] = myw[q];
    }
    __syncthreads();
#pragma unroll
    for (int q = 0; q < 4; ++q) {
        int e = threadIdx.x + 256 * q;
        int i = e >> 4, j = e & 15;
        float acc = 0.f;
        for (int k = 0; k < 64; ++k) acc += cs[i][k] * wsld[k][j];
        atomicAdd(&s2s[j], myw[q] * acc);
        w0[(size_t)g * 1024 + e] = myw[q];
    }
    __syncthreads();
    if (threadIdx.x < 16) s2[(size_t)g * 16 + threadIdx.x] = s2s[threadIdx.x];
}

// ----------------------------------------------------------------------------
// p[g][j] = softmax_g( -(t*yc2[j] - s2[g][j] + S[g]) )
// ----------------------------------------------------------------------------
__global__ void k_softmax(const float* __restrict__ s2, const float* __restrict__ Sg,
                          const float* __restrict__ yc2, const float* __restrict__ sc,
                          float* __restrict__ p) {
    __shared__ float lg[64][16];
    float tpar = sc[1];
    for (int e = threadIdx.x; e < 64 * 16; e += 256) {
        int g = e >> 4, j = e & 15;
        lg[g][j] = -(tpar * yc2[j] - s2[(size_t)g * 16 + j] + Sg[g]);
    }
    __syncthreads();
    if (threadIdx.x < 16) {
        int j = threadIdx.x;
        float m = -1e30f;
        for (int g = 0; g < 64; ++g) m = fmaxf(m, lg[g][j]);
        float s = 0.f;
        for (int g = 0; g < 64; ++g) s += expf(lg[g][j] - m);
        for (int g = 0; g < 64; ++g) p[(size_t)g * 16 + j] = expf(lg[g][j] - m) / s;
    }
}

// wmat[f][j] = w0[f][j] * p[f/64][j]   (flat index f*16+j)
__global__ void k_wscale(const float* __restrict__ w0, const float* __restrict__ p,
                         float* __restrict__ wmat, size_t n) {
    size_t i = (size_t)blockIdx.x * blockDim.x + threadIdx.x;
    if (i >= n) return;
    size_t g = i / 1024;           // 64*16
    wmat[i] = w0[i] * p[g * 16 + (i & 15)];
}

// qy[n][j] = Xq[n][:] . wmat[:][j] + b[j]
__global__ void k_qy(const float* __restrict__ Xq, const float* __restrict__ wmat,
                     const float* __restrict__ bvec, float* __restrict__ qy,
                     int N, int C, int NY) {
    size_t id = (size_t)blockIdx.x * blockDim.x + threadIdx.x;
    if (id >= (size_t)N * NY) return;
    int j = (int)(id % NY);
    int n = (int)(id / NY);
    float acc = bvec[j];
    for (int f = 0; f < C; ++f)
        acc += Xq[(size_t)n * C + f] * wmat[(size_t)f * NY + j];
    qy[id] = acc;
}

// Yvar[n][g] = x_g^T cov[g] x_g ,  x_g = Xq[n, g*64:+64]
__global__ __launch_bounds__(64) void k_yvar(const float* __restrict__ Xq,
                                             const float* __restrict__ cov,
                                             float* __restrict__ Yvar, int C) {
    int n = blockIdx.x;
    int i = threadIdx.x;
    __shared__ float xg[64];
    __shared__ float part[64];
    for (int g = 0; g < 64; ++g) {
        __syncthreads();
        xg[i] = Xq[(size_t)n * C + g * 64 + i];
        __syncthreads();
        const float* cr = cov + (size_t)g * 4096 + (size_t)i * 64;
        float acc = 0.f;
        for (int k = 0; k < 64; ++k) acc += cr[k] * xg[k];
        part[i] = acc * xg[i];
        __syncthreads();
        for (int s = 32; s > 0; s >>= 1) {
            if (i < s) part[i] += part[i + s];
            __syncthreads();
        }
        if (i == 0) Yvar[(size_t)n * 64 + g] = part[0];
    }
}

// out[:N*NY] = qy*ymul ; out[N*NY:] = sqrt(max(sum_g Yvar*p, 0))*ymul
__global__ void k_final(const float* __restrict__ qy, const float* __restrict__ Yvar,
                        const float* __restrict__ p, const float* __restrict__ ymul,
                        float* __restrict__ out, int N, int NY) {
    size_t id = (size_t)blockIdx.x * blockDim.x + threadIdx.x;
    if (id >= (size_t)N * NY) return;
    int j = (int)(id % NY);
    int n = (int)(id / NY);
    float ym = ymul[j];
    float acc = 0.f;
    for (int g = 0; g < 64; ++g)
        acc += Yvar[(size_t)n * 64 + g] * p[(size_t)g * 16 + j];
    out[id] = qy[id] * ym;
    out[(size_t)N * NY + id] = sqrtf(fmaxf(acc, 0.f)) * ym;
}

// ----------------------------------------------------------------------------
extern "C" void kernel_launch(void* const* d_in, const int* in_sizes, int n_in,
                              void* d_out, int out_size, void* d_ws, size_t ws_size,
                              hipStream_t stream) {
    (void)in_sizes; (void)n_in; (void)out_size; (void)ws_size;
    const int*   x    = (const int*)d_in[0];
    const float* y    = (const float*)d_in[1];
    const int*   qx   = (const int*)d_in[2];
    const float* we   = (const float*)d_in[3];
    const float* W1   = (const float*)d_in[4];
    const float* b1   = (const float*)d_in[5];
    const float* W2   = (const float*)d_in[6];
    const float* b2   = (const float*)d_in[7];
    const float* W3   = (const float*)d_in[8];
    const float* b3   = (const float*)d_in[9];
    const float* regp = (const float*)d_in[10];
    const float* tp   = (const float*)d_in[11];

    constexpr int Nn = 2048, NQ = 2048, NY = 16, L = 8, D = 768;
    constexpr int NIN = L * D;            // 6144
    constexpr int Hh = 512, NH = 4096;
    constexpr int Gg = 64;

    char* ws = (char*)d_ws;
    size_t off = 0;
    auto alloc = [&](size_t bytes) -> char* {
        char* ptr = ws + off;
        off += (bytes + 255) & ~(size_t)255;
        return ptr;
    };

    float*    sc   = (float*)alloc(2 * sizeof(float));
    _Float16* W1h  = (_Float16*)alloc((size_t)Hh * NIN * 2);
    _Float16* W2h  = (_Float16*)alloc((size_t)Hh * Hh * 2);
    _Float16* W3h  = (_Float16*)alloc((size_t)NH * Hh * 2);
    _Float16* Ein  = (_Float16*)alloc((size_t)Nn * NIN * 2);
    _Float16* H1h  = (_Float16*)alloc((size_t)Nn * Hh * 2);
    _Float16* H2h  = (_Float16*)alloc((size_t)Nn * Hh * 2);
    float*    E    = (float*)alloc((size_t)Nn * NH * 4);   // Ex, then reused as Eq
    float*    xmul = (float*)alloc(NH * 4);
    float*    ymul = (float*)alloc(NY * 4);
    float*    bvec = (float*)alloc(NY * 4);
    float*    yc2  = (float*)alloc(NY * 4);
    float*    amat = (float*)alloc((size_t)NH * NY * 4);
    float*    covi = (float*)alloc((size_t)Gg * 64 * 64 * 4);
    float*    cov  = (float*)alloc((size_t)Gg * 64 * 64 * 4);
    float*    Sg   = (float*)alloc(Gg * 4);
    float*    w0   = (float*)alloc((size_t)Gg * 64 * NY * 4);
    float*    s2   = (float*)alloc((size_t)Gg * NY * 4);
    float*    pgj  = (float*)alloc((size_t)Gg * NY * 4);
    float*    wmat = (float*)alloc((size_t)NH * NY * 4);
    float*    qy   = (float*)alloc((size_t)NQ * NY * 4);
    float*    Yvar = (float*)alloc((size_t)NQ * Gg * 4);

    float* out = (float*)d_out;
    auto blks = [](size_t n) { return (unsigned)((n + 255) / 256); };

    k_scalars<<<1, 32, 0, stream>>>(regp, tp, sc);
    { size_t n = (size_t)Hh * NIN; k_f32_to_f16<<<blks(n), 256, 0, stream>>>(W1, W1h, n); }
    { size_t n = (size_t)Hh * Hh;  k_f32_to_f16<<<blks(n), 256, 0, stream>>>(W2, W2h, n); }
    { size_t n = (size_t)NH * Hh;  k_f32_to_f16<<<blks(n), 256, 0, stream>>>(W3, W3h, n); }

    auto embed = [&](const int* tok) {
        size_t n = (size_t)Nn * NIN;
        k_gather<<<blks(n), 256, 0, stream>>>(tok, we, Ein, Nn, L, D);
        k_gemm_nt<true ><<<dim3(Nn / 64, Hh / 128), 256, 0, stream>>>(Ein, W1h, b1, H1h, nullptr, Nn, Hh, NIN);
        k_gemm_nt<true ><<<dim3(Nn / 64, Hh / 128), 256, 0, stream>>>(H1h, W2h, b2, H2h, nullptr, Nn, Hh, Hh);
        k_gemm_nt<false><<<dim3(Nn / 64, NH / 128), 256, 0, stream>>>(H2h, W3h, b3, nullptr, E, Nn, NH, Hh);
    };

    // -------- ridge_learn on embed(x) --------
    embed(x);
    k_colstd<<<NH / 256, 256, 0, stream>>>(E, xmul, Nn, NH);
    { size_t n = (size_t)Nn * NH; k_scale_cols<<<blks(n), 256, 0, stream>>>(E, xmul, Nn, NH); }
    k_ystats<<<1, 256, 0, stream>>>(y, ymul, bvec, yc2, Nn, NY);
    { size_t n = (size_t)NH * NY; k_amat<<<blks(n), 256, 0, stream>>>(E, y, ymul, bvec, amat, Nn, NH, NY); }
    k_cov<<<Gg, 256, 0, stream>>>(E, sc, covi, Nn, NH);
    k_inv<<<Gg, 64, 0, stream>>>(covi, cov, Sg);
    k_w0<<<Gg, 256, 0, stream>>>(cov, amat, sc, w0, s2);
    k_softmax<<<1, 256, 0, stream>>>(s2, Sg, yc2, sc, pgj);
    { size_t n = (size_t)NH * NY; k_wscale<<<blks(n), 256, 0, stream>>>(w0, pgj, wmat, n); }

    // -------- ridge_predict on embed(qx) (E buffer reused) --------
    embed(qx);
    { size_t n = (size_t)NQ * NH; k_scale_cols<<<blks(n), 256, 0, stream>>>(E, xmul, NQ, NH); }
    { size_t n = (size_t)NQ * NY; k_qy<<<blks(n), 256, 0, stream>>>(E, wmat, bvec, qy, NQ, NH, NY); }
    k_yvar<<<NQ, 64, 0, stream>>>(E, cov, Yvar, NH);
    { size_t n = (size_t)NQ * NY; k_final<<<blks(n), 256, 0, stream>>>(qy, Yvar, pgj, ymul, out, NQ, NY); }
}
